// EMAQuantizer_52544629899291
// MI455X (gfx1250) — matrix operand order repaired and checked by reading, not compiled, not run on hardware
//
#include <hip/hip_runtime.h>
#include <hip/hip_bf16.h>
#include <math.h>

// ---------------------------------------------------------------------------
// EMA vector quantizer for MI455X (gfx1250, wave32, WMMA).
// Dominant work: 32768x8192x64 distance GEMM -> v_wmma_f32_16x16x32_bf16
// with bf16 hi/lo split (3 products) for near-fp32 accuracy.
// v2: 2x M register blocking per wave (32 rows) -> halves L2 codebook
// traffic (4 GB -> 2 GB) and gives 12 WMMAs per 8 b128 loads, with two
// independent accumulator chains for ILP.
// ---------------------------------------------------------------------------

typedef __bf16 bf16_t;
typedef __attribute__((ext_vector_type(16))) __bf16 v16bf;
typedef __attribute__((ext_vector_type(8)))  float  v8f;

#define VQ_NUM_CODES 8192
#define VQ_CODE_DIM  64
#define VQ_N         32768      // B*H*W = 8*64*64
#define VQ_DECAY     0.99f
#define VQ_EPS       1e-5f

union FragU { v16bf v; uint4 q[2]; };

static __device__ __forceinline__ v8f wmma_bf16(const v16bf& a, const v16bf& b, v8f c) {
    // D = A(16x32 bf16) x B(32x16 bf16) + C(16x16 f32)
    return __builtin_amdgcn_wmma_f32_16x16x32_bf16(
        /*neg_a=*/false, a, /*neg_b=*/false, b,
        /*c_mod=*/(short)0, c, /*reuse_a=*/false, /*reuse_b=*/false);
}

// ---------------------------------------------------------------------------
// Kernel 1: split embed (8192x64 f32) into bf16 hi/lo planes + row norms e2.
// ---------------------------------------------------------------------------
__global__ void vq_prep(const float* __restrict__ embed,
                        bf16_t* __restrict__ ehi, bf16_t* __restrict__ elo,
                        float* __restrict__ e2) {
    int j = blockIdx.x, c = threadIdx.x;
    float x = embed[j * VQ_CODE_DIM + c];
    bf16_t h = (bf16_t)x;
    ehi[j * VQ_CODE_DIM + c] = h;
    elo[j * VQ_CODE_DIM + c] = (bf16_t)(x - (float)h);
    __shared__ float s[VQ_CODE_DIM];
    s[c] = x * x;
    __syncthreads();
    for (int o = 32; o > 0; o >>= 1) {
        if (c < o) s[c] += s[c + o];
        __syncthreads();
    }
    if (c == 0) e2[j] = s[0];
}

// Zero the atomic-accumulation scratch (cs_batch + embed_sum + nsum).
__global__ void vq_zero(float* __restrict__ p, int n) {
    int i = blockIdx.x * blockDim.x + threadIdx.x;
    if (i < n) p[i] = 0.0f;
}

// ---------------------------------------------------------------------------
// Kernel 2: WMMA distance GEMM + argmin.
// Block = 256 threads = 8 waves; each wave owns TWO 16-row tiles of "flat"
// (flat[n][c] = z[b][c][h][w], n = b*4096 + h*64 + w) and sweeps all 512
// 16-code tiles. Each B fragment feeds two independent 6-WMMA chains.
// argmin key: e2[j] - 2*dot (row term x2 is constant per row, dropped).
// ---------------------------------------------------------------------------
__global__ void __launch_bounds__(256)
vq_assign(const float* __restrict__ z,
          const bf16_t* __restrict__ ehi, const bf16_t* __restrict__ elo,
          const float* __restrict__ e2,
          int* __restrict__ codes) {
    __shared__ bf16_t aHi[8][32 * VQ_CODE_DIM];   // 32 KB
    __shared__ bf16_t aLo[8][32 * VQ_CODE_DIM];   // 32 KB
    __shared__ float  rmin[8][32][16];            // 16 KB
    __shared__ int    ridx[8][32][16];            // 16 KB

    const int lane = threadIdx.x & 31;
    const int wave = threadIdx.x >> 5;
    const int n0   = blockIdx.x * 256 + wave * 32;

    // ---- Stage A tiles (32 rows x 64 dims) into LDS as bf16 hi/lo.
    // i = k*32 + lane, m = i&31, c = i>>5 -> 32 consecutive hw per load.
    for (int i = lane; i < 32 * VQ_CODE_DIM; i += 32) {
        int m = i & 31, c = i >> 5;
        int n = n0 + m;
        int b = n >> 12, hw = n & 4095;
        float x = z[((b * VQ_CODE_DIM + c) << 12) + hw];
        bf16_t h = (bf16_t)x;
        aHi[wave][(m << 6) + c] = h;
        aLo[wave][(m << 6) + c] = (bf16_t)(x - (float)h);
    }
    __syncthreads();

    // ---- A fragments (kept in VGPRs for the whole code sweep).
    // 16-bit A 16x32 layout: lane M = lane%16, VGPR v: K = (v>=4)*16 + g*8 + (v%4)*2
    const int g = lane >> 4;
    const int m = lane & 15;
    FragU ah0[2], ah1[2], al0[2], al1[2];
#pragma unroll
    for (int t = 0; t < 2; ++t) {
        const bf16_t* arowh = &aHi[wave][(t * 16 + m) << 6];
        const bf16_t* arowl = &aLo[wave][(t * 16 + m) << 6];
        ah0[t].q[0] = *(const uint4*)(arowh + g * 8);       // K  0..7 / 8..15
        ah0[t].q[1] = *(const uint4*)(arowh + 16 + g * 8);  // K 16..23 / 24..31
        ah1[t].q[0] = *(const uint4*)(arowh + 32 + g * 8);
        ah1[t].q[1] = *(const uint4*)(arowh + 48 + g * 8);
        al0[t].q[0] = *(const uint4*)(arowl + g * 8);
        al0[t].q[1] = *(const uint4*)(arowl + 16 + g * 8);
        al1[t].q[0] = *(const uint4*)(arowl + 32 + g * 8);
        al1[t].q[1] = *(const uint4*)(arowl + 48 + g * 8);
    }

    float best[2][8];
    int   bidx[2][8];
#pragma unroll
    for (int t = 0; t < 2; ++t)
#pragma unroll
        for (int r = 0; r < 8; ++r) { best[t][r] = 3.4e38f; bidx[t][r] = 0; }

    // ---- Sweep all 8192 codes, 16 at a time (L2-resident bf16 codebook).
    for (int j = 0; j < VQ_NUM_CODES; j += 16) {
        const int jn = j + m;                            // this lane's column/code
        const bf16_t* bh = ehi + jn * VQ_CODE_DIM;
        const bf16_t* bl = elo + jn * VQ_CODE_DIM;
        // B 32x16 layout: lane N = lane%16, VGPR v: K = (v>=4)*16 + g*8 + (v%4)*2 (+1)
        FragU bh0, bh1, bl0, bl1;
        bh0.q[0] = *(const uint4*)(bh + g * 8);        bh0.q[1] = *(const uint4*)(bh + 16 + g * 8);
        bh1.q[0] = *(const uint4*)(bh + 32 + g * 8);   bh1.q[1] = *(const uint4*)(bh + 48 + g * 8);
        bl0.q[0] = *(const uint4*)(bl + g * 8);        bl0.q[1] = *(const uint4*)(bl + 16 + g * 8);
        bl1.q[0] = *(const uint4*)(bl + 32 + g * 8);   bl1.q[1] = *(const uint4*)(bl + 48 + g * 8);
        const float ee = e2[jn];

        v8f acc[2];
#pragma unroll
        for (int t = 0; t < 2; ++t) {
            v8f a = {};
            a = wmma_bf16(ah0[t].v, bh0.v, a);   // hi*hi  (K 0..31)
            a = wmma_bf16(al0[t].v, bh0.v, a);   // lo*hi
            a = wmma_bf16(ah0[t].v, bl0.v, a);   // hi*lo
            a = wmma_bf16(ah1[t].v, bh1.v, a);   // hi*hi  (K 32..63)
            a = wmma_bf16(al1[t].v, bh1.v, a);
            a = wmma_bf16(ah1[t].v, bl1.v, a);
            acc[t] = a;
        }

#pragma unroll
        for (int t = 0; t < 2; ++t)
#pragma unroll
            for (int r = 0; r < 8; ++r) {
                float d = ee - 2.0f * acc[t][r];
                if (d < best[t][r]) { best[t][r] = d; bidx[t][r] = jn; }
            }
    }

    // ---- Cross-lane argmin per row (C layout: VGPR r -> M = r + 8*g, N = lane%16).
    __syncthreads();
#pragma unroll
    for (int t = 0; t < 2; ++t)
#pragma unroll
        for (int r = 0; r < 8; ++r) {
            int mm = t * 16 + r + (g << 3);
            rmin[wave][mm][m] = best[t][r];
            ridx[wave][mm][m] = bidx[t][r];
        }
    __syncthreads();
    {
        // all 32 lanes: one output row each
        float bv = rmin[wave][lane][0];
        int   bi = ridx[wave][lane][0];
        for (int u = 1; u < 16; ++u) {
            float v = rmin[wave][lane][u];
            int  ii = ridx[wave][lane][u];
            if (v < bv || (v == bv && ii < bi)) { bv = v; bi = ii; }
        }
        codes[n0 + lane] = bi;
    }
}

// ---------------------------------------------------------------------------
// Segment sums via f32 atomics (L2 resident, ~2.1M atomic adds total).
// ---------------------------------------------------------------------------
__global__ void vq_count(const int* __restrict__ codes, float* __restrict__ cs_batch) {
    int n = blockIdx.x * blockDim.x + threadIdx.x;
    if (n < VQ_N) atomicAdd(&cs_batch[codes[n]], 1.0f);
}

__global__ void vq_scatter(const float* __restrict__ z, const int* __restrict__ codes,
                           float* __restrict__ embed_sum) {
    int t = blockIdx.x * blockDim.x + threadIdx.x;   // 2M threads, hw-fast
    int c = t >> 15;
    int n = t & (VQ_N - 1);
    int b = n >> 12, hw = n & 4095;
    float x = z[((b * VQ_CODE_DIM + c) << 12) + hw];
    atomicAdd(&embed_sum[codes[n] * VQ_CODE_DIM + c], x);
}

// ---------------------------------------------------------------------------
// EMA cluster size + global sum n (single block).
// ---------------------------------------------------------------------------
__global__ void vq_reduce(const float* __restrict__ cluster_size,
                          const float* __restrict__ cs_batch,
                          float* __restrict__ new_cs_out, float* __restrict__ nsum) {
    __shared__ float s[256];
    float acc = 0.0f;
    for (int i = threadIdx.x; i < VQ_NUM_CODES; i += 256) {
        float v = cluster_size[i] * VQ_DECAY + cs_batch[i] * (1.0f - VQ_DECAY);
        new_cs_out[i] = v;
        acc += v;
    }
    s[threadIdx.x] = acc;
    __syncthreads();
    for (int o = 128; o > 0; o >>= 1) {
        if (threadIdx.x < o) s[threadIdx.x] += s[threadIdx.x + o];
        __syncthreads();
    }
    if (threadIdx.x == 0) *nsum = s[0];
}

// ---------------------------------------------------------------------------
// EMA embed_avg + Laplace-smoothed normalize -> new_embed.
// ---------------------------------------------------------------------------
__global__ void vq_final(const float* __restrict__ embed_avg,
                         const float* __restrict__ embed_sum,
                         const float* __restrict__ new_cs,
                         const float* __restrict__ nsum,
                         float* __restrict__ new_embed_out,
                         float* __restrict__ new_ea_out) {
    int j = blockIdx.x, c = threadIdx.x;
    float nea = embed_avg[j * VQ_CODE_DIM + c] * VQ_DECAY +
                embed_sum[j * VQ_CODE_DIM + c] * (1.0f - VQ_DECAY);
    new_ea_out[j * VQ_CODE_DIM + c] = nea;
    float nv = *nsum;
    float denom = (new_cs[j] + VQ_EPS) / (nv + (float)VQ_NUM_CODES * VQ_EPS) * nv;
    float e = nea / denom;
    __shared__ float s[VQ_CODE_DIM];
    s[c] = e * e;
    __syncthreads();
    for (int o = 32; o > 0; o >>= 1) {
        if (c < o) s[c] += s[c + o];
        __syncthreads();
    }
    new_embed_out[j * VQ_CODE_DIM + c] = e * rsqrtf(s[0]);
}

// ---------------------------------------------------------------------------
// z_q gather (old embed) + codes output (float-cast indices).
// ---------------------------------------------------------------------------
__global__ void vq_zq(const float* __restrict__ embed, const int* __restrict__ codes,
                      float* __restrict__ zq, float* __restrict__ codes_f) {
    int t = blockIdx.x * blockDim.x + threadIdx.x;   // t == (b*64+c)*4096 + hw
    int hw = t & 4095;
    int bc = t >> 12;
    int c = bc & 63, b = bc >> 6;
    int n = (b << 12) + hw;
    int code = codes[n];
    zq[t] = embed[code * VQ_CODE_DIM + c];
    if (c == 0) codes_f[n] = (float)code;
}

// ---------------------------------------------------------------------------
extern "C" void kernel_launch(void* const* d_in, const int* in_sizes, int n_in,
                              void* d_out, int out_size, void* d_ws, size_t ws_size,
                              hipStream_t stream) {
    const float* z            = (const float*)d_in[0];   // 8x64x64x64
    const float* embed        = (const float*)d_in[1];   // 8192x64
    const float* cluster_size = (const float*)d_in[2];   // 8192
    const float* embed_avg    = (const float*)d_in[3];   // 8192x64

    // ---- workspace layout (bytes)
    char* ws = (char*)d_ws;
    bf16_t* ehi      = (bf16_t*)(ws);                              // 1 MB
    bf16_t* elo      = (bf16_t*)(ws + 1048576);                    // 1 MB
    float*  e2       = (float*)(ws + 2097152);                     // 32 KB
    int*    codes_i  = (int*)(ws + 2129920);                       // 128 KB
    float*  cs_batch = (float*)(ws + 2260992);                     // 32 KB
    float*  esum     = (float*)(ws + 2293760);                     // 2 MB
    float*  nsum     = (float*)(ws + 4390912);                     // 4 B

    // ---- output layout (floats), tuple order:
    float* out       = (float*)d_out;
    float* o_zq      = out;                    // 2097152
    float* o_codes   = out + 2097152;          //   32768
    float* o_embed   = out + 2129920;          //  524288
    float* o_cs      = out + 2654208;          //    8192
    float* o_ea      = out + 2662400;          //  524288

    // 1) codebook -> bf16 hi/lo + row norms
    vq_prep<<<VQ_NUM_CODES, VQ_CODE_DIM, 0, stream>>>(embed, ehi, elo, e2);
    // 2) zero atomic scratch (cs_batch 8192 + esum 524288 + nsum 1)
    vq_zero<<<(532481 + 255) / 256, 256, 0, stream>>>(cs_batch, 532481);
    // 3) WMMA distance GEMM + argmin (128 blocks x 8 waves x 32 rows)
    vq_assign<<<128, 256, 0, stream>>>(z, ehi, elo, e2, codes_i);
    // 4) segment sums
    vq_count<<<(VQ_N + 255) / 256, 256, 0, stream>>>(codes_i, cs_batch);
    vq_scatter<<<(VQ_N * VQ_CODE_DIM) / 256, 256, 0, stream>>>(z, codes_i, esum);
    // 5) EMA cluster size + total
    vq_reduce<<<1, 256, 0, stream>>>(cluster_size, cs_batch, o_cs, nsum);
    // 6) EMA embed_avg + normalized new embed
    vq_final<<<VQ_NUM_CODES, VQ_CODE_DIM, 0, stream>>>(embed_avg, esum, o_cs, nsum,
                                                       o_embed, o_ea);
    // 7) z_q gather (uses OLD embed) + codes
    vq_zq<<<(VQ_N * VQ_CODE_DIM) / 256, 256, 0, stream>>>(embed, codes_i, o_zq, o_codes);
}
